// MHParallelAttention_27419071217846
// MI455X (gfx1250) — matrix operand order repaired and checked by reading, not compiled, not run on hardware
//
#include <hip/hip_runtime.h>
#include <hip/hip_bf16.h>

// MHParallelAttention for MI455X (gfx1250, wave32).
//
// comb[b,q,k] = sum_h Wc[h] * (tanh(Q Wq^T+bq)_h . tanh(K Wk^T+bk)_h)
//             = Qbig[b,q,:] . Kbig[b,k,:]      (Wc folded into Kbig)
// Stage 2: f16 WMMA GEMM with TDM (tensor_load_to_lds) double-buffered B
// staging per wave, fused mask + softmax.

typedef _Float16 v16h __attribute__((ext_vector_type(16)));
typedef _Float16 v8h  __attribute__((ext_vector_type(8)));
typedef float    v8f  __attribute__((ext_vector_type(8)));
typedef unsigned int u32x4 __attribute__((ext_vector_type(4)));
typedef int          i32x8 __attribute__((ext_vector_type(8)));
typedef int          i32x4 __attribute__((ext_vector_type(4)));

union Frag16 { v16h v; v8h h[2]; };

#define NEGV (-10000000000.0f)

#if defined(__has_builtin)
#if __has_builtin(__builtin_amdgcn_tensor_load_to_lds) && \
    __has_builtin(__builtin_amdgcn_s_wait_tensorcnt)
#define USE_TDM 1
#endif
#endif
#ifndef USE_TDM
#define USE_TDM 0
#endif

// ---------------------------------------------------------------------------
// Stage 1: per-head Linear + tanh, fold Wc into K side, emit f16 Qbig/Kbig.
// grid = (B*1024, 2): x -> (b, s2) row, y -> 0:Q 1:K. block = 256.
// ---------------------------------------------------------------------------
__global__ __launch_bounds__(256) void proj_tanh_kernel(
    const float* __restrict__ query, const float* __restrict__ keyin,
    const float* __restrict__ Wq, const float* __restrict__ bq,
    const float* __restrict__ Wk, const float* __restrict__ bk,
    const float* __restrict__ Wc,
    _Float16* __restrict__ Qbig, _Float16* __restrict__ Kbig)
{
    __shared__ float Ws[64 * 65];   // 65-stride: conflict-free column reads
    __shared__ float inS[16 * 64];  // the 16 input rows this (b,s2) touches
    __shared__ float biasS[64];
    __shared__ float WcS[16];

    const int tid = threadIdx.x;
    const int isK = blockIdx.y;
    const int b   = blockIdx.x >> 10;
    const int s2  = blockIdx.x & 1023;

    const float* src  = isK ? keyin : query;
    const float* W    = isK ? Wk : Wq;
    const float* bias = isK ? bk : bq;

    #pragma unroll
    for (int i = tid; i < 4096; i += 256) {
        int e = i >> 6, d = i & 63;
        Ws[e * 65 + d] = W[i];
    }
    // reshape(B,H,-1,DK): head h, row s2 <- query[b, h*64+(s2>>4), (s2&15)*64+d]
    for (int i = tid; i < 1024; i += 256) {
        int h = i >> 6, d = i & 63;
        inS[i] = src[((size_t)(b * 1024 + h * 64 + (s2 >> 4))) * 1024 + (s2 & 15) * 64 + d];
    }
    if (tid < 64) biasS[tid] = bias[tid];
    if (tid < 16) WcS[tid]   = Wc[tid];
    __syncthreads();

    _Float16* dst = isK ? Kbig : Qbig;
    #pragma unroll
    for (int j = 0; j < 4; j++) {
        int col = tid + j * 256;        // col = h*64 + e
        int h = col >> 6, e = col & 63;
        const float* inrow = &inS[h * 64];
        const float* wrow  = &Ws[e * 65];
        float acc = biasS[e];
        #pragma unroll
        for (int d = 0; d < 64; d++) acc += inrow[d] * wrow[d];
        float t = tanhf(acc);
        if (isK) t *= WcS[h];           // fold head-combination weight into K
        dst[((size_t)(b * 1024 + s2)) * 1024 + col] = (_Float16)t;
    }
}

// ---------------------------------------------------------------------------
// TDM: DMA a 128-row x 32-half tile of Kbig into LDS (per-wave, private).
// Padding: 4 DWORDs (16 B) after every 16 DWORDs (64 B = one tile row)
// -> 80 B LDS stride per column, near-conflict-free ds_load_b128 reads.
// ---------------------------------------------------------------------------
__device__ __forceinline__ void tdm_load_b_tile(unsigned ldsOff,
                                                unsigned long long gaddr) {
#if USE_TDM
    u32x4 g0;
    g0[0] = 1u;                                      // count=1 (valid D#)
    g0[1] = ldsOff;                                  // lds_addr (bytes)
    g0[2] = (unsigned)gaddr;                         // global_addr[31:0]
    g0[3] = (unsigned)(gaddr >> 32) | (2u << 30);    // global_addr[56:32]|type=2
    i32x8 g1;
    g1[0] = (1 << 16)            // data_size = 2 bytes
          | (1 << 20)            // pad_enable
          | (3 << 22)            // pad_interval: 16 DWORDs (= 64 B tile row)
          | (3 << 25);           // pad_amount: 4 DWORDs (16 B)
    g1[1] = (int)(1024u << 16);  // tensor_dim0 = 1024 (K length)
    g1[2] = (int)(1024u << 16);  // tensor_dim1 = 1024 (rows)
    g1[3] = (int)(32u << 16);    // tile_dim0 = 32 (K halves)
    g1[4] = 128;                 // tile_dim1 = 128 rows, tile_dim2 = 0
    g1[5] = 1024;                // tensor_dim0_stride = 1024 elements
    g1[6] = 0;
    g1[7] = 0;
    i32x4 z4 = {0, 0, 0, 0};               // groups 2/3 unused for 2D tile
    i32x8 z8 = {0, 0, 0, 0, 0, 0, 0, 0};   // extra group (6-arg toolchain form)
    __builtin_amdgcn_tensor_load_to_lds(g0, g1, z4, z4, z8, 0);
#else
    (void)ldsOff; (void)gaddr;
#endif
}

// ---------------------------------------------------------------------------
// Stage 2: comb = Qbig @ Kbig^T (+bc, mask) then row softmax.
// grid = (64, 4): x -> 16-row q block, y -> batch. block = 256 (8 waves).
// Per wave: 16(M) x 128(N), K-loop of 32 steps (K=32 each).
// LDS (dynamic 160KB): per-wave double-buffered B tiles; comb tile aliased.
// ---------------------------------------------------------------------------
__global__ __launch_bounds__(256) void attn_softmax_kernel(
    const _Float16* __restrict__ Qbig, const _Float16* __restrict__ Kbig,
    const int* __restrict__ mask, const float* __restrict__ bc,
    float* __restrict__ out)
{
    extern __shared__ __attribute__((aligned(16))) _Float16 smem[];

    const int tid   = threadIdx.x;
    const int wave  = __builtin_amdgcn_readfirstlane(tid >> 5);  // SGPR-uniform
    const int lane  = tid & 31;
    const int b     = blockIdx.y;
    const int qbase = blockIdx.x * 16;
    const int nbase = wave * 128;
    const int lm    = lane & 15;       // row (A) / col (B) within 16x16 tile
    const int lh    = lane >> 4;       // half-wave selector

    v8f acc[8];
    const v8f vzero = {0.f, 0.f, 0.f, 0.f, 0.f, 0.f, 0.f, 0.f};
    #pragma unroll
    for (int t = 0; t < 8; t++) acc[t] = vzero;

    // A 16x32 f16 layout: lane holds row lm, contiguous K chunk [lh*16,+16).
    const _Float16* Arow =
        Qbig + ((size_t)(b * 1024 + qbase + lm)) * 1024 + lh * 16;
    Frag16 a[2];
    a[0].h[0] = *(const v8h*)(Arow);
    a[0].h[1] = *(const v8h*)(Arow + 8);

#if USE_TDM
    const unsigned bufB = (unsigned)wave * 20480u;   // 2 x 10240 B per wave
    const unsigned long long kBase =
        (unsigned long long)(uintptr_t)Kbig +
        ((unsigned long long)(b * 1024 + nbase)) * 2048ull;
    tdm_load_b_tile(bufB,          kBase);        // stage 0 (kk =  0)
    tdm_load_b_tile(bufB + 10240u, kBase + 64u);  // stage 1 (kk = 32)
#endif

    for (int ks = 0; ks < 32; ks++) {
        const int kk  = ks * 32;
        const int cur = ks & 1;
        if (ks < 31) {   // register double-buffer of next A fragment
            a[cur ^ 1].h[0] = *(const v8h*)(Arow + kk + 32);
            a[cur ^ 1].h[1] = *(const v8h*)(Arow + kk + 40);
        }
#if USE_TDM
        if (ks == 31) __builtin_amdgcn_s_wait_tensorcnt(0);
        else          __builtin_amdgcn_s_wait_tensorcnt(1);
        // B 32x16 frag: lane holds col lm, contiguous K chunk [lh*16,+16).
        const _Float16* bbuf = smem + (bufB + (unsigned)cur * 10240u) / 2;
        #pragma unroll
        for (int t = 0; t < 8; t++) {
            const _Float16* bp = bbuf + (t * 16 + lm) * 40 + lh * 16;
            Frag16 bf;
            bf.h[0] = *(const v8h*)bp;
            bf.h[1] = *(const v8h*)(bp + 8);
            acc[t] = __builtin_amdgcn_wmma_f32_16x16x32_f16(
                false, a[cur].v, false, bf.v, (short)0, acc[t], false, false);
        }
        // prefetch stage ks+2 into the buffer just consumed (ds reads retired
        // by the wmma dscnt waits above, so overwrite is safe)
        if (ks < 30)
            tdm_load_b_tile(bufB + (unsigned)cur * 10240u,
                            kBase + (unsigned long long)(kk + 64) * 2ull);
#else
        Frag16 bf[8];
        #pragma unroll
        for (int t = 0; t < 8; t++) {
            int col = nbase + t * 16 + lm;
            const v8h* pb = (const v8h*)(
                Kbig + ((size_t)(b * 1024 + col)) * 1024 + kk + lh * 16);
            bf[t].h[0] = pb[0];
            bf[t].h[1] = pb[1];
        }
        #pragma unroll
        for (int t = 0; t < 8; t++)
            acc[t] = __builtin_amdgcn_wmma_f32_16x16x32_f16(
                false, a[cur].v, false, bf[t].v, (short)0, acc[t], false, false);
#endif
    }

    __syncthreads();   // all waves done with B buffers; alias comb tile on top
    float* lds = (float*)smem;   // 16 x 1024 f32 comb tile (64 KB)

    // C/D layout: VGPR r, lanes 0-15 -> (M=r, N=lm); lanes 16-31 -> (M=8+r).
    #pragma unroll
    for (int t = 0; t < 8; t++) {
        #pragma unroll
        for (int r = 0; r < 8; r++) {
            int m   = r + lh * 8;
            int col = nbase + t * 16 + lm;
            lds[m * 1024 + col] = acc[t][r];
        }
    }
    __syncthreads();

    // Row softmax: 16 threads per row, 64 columns each (stride 16).
    const int row = tid >> 4;
    const int c   = tid & 15;
    const float bc0 = bc[0];
    const int* mrow = mask + ((size_t)(b * 1024 + qbase + row)) * 1024;

    float vals[64];
    float maxv = -3.4e38f;
    #pragma unroll
    for (int j = 0; j < 64; j++) {
        int col = c + j * 16;
        float v = lds[row * 1024 + col] + bc0;
        if (mrow[col] == 0) v = NEGV;
        vals[j] = v;
        maxv = fmaxf(maxv, v);
    }
    #pragma unroll
    for (int m = 1; m < 16; m <<= 1) maxv = fmaxf(maxv, __shfl_xor(maxv, m, 32));

    float sum = 0.f;
    #pragma unroll
    for (int j = 0; j < 64; j++) {
        float e = __expf(vals[j] - maxv);
        vals[j] = e;
        sum += e;
    }
    #pragma unroll
    for (int m = 1; m < 16; m <<= 1) sum += __shfl_xor(sum, m, 32);

    const float inv = 1.0f / sum;
    float* orow = out + ((size_t)(b * 1024 + qbase + row)) * 1024;
    #pragma unroll
    for (int j = 0; j < 64; j++) orow[c + j * 16] = vals[j] * inv;
}

// ---------------------------------------------------------------------------
extern "C" void kernel_launch(void* const* d_in, const int* in_sizes, int n_in,
                              void* d_out, int out_size, void* d_ws, size_t ws_size,
                              hipStream_t stream) {
    const float* query = (const float*)d_in[0];
    const float* keyin = (const float*)d_in[1];
    const int*   mask  = (const int*)d_in[2];
    const float* Wq    = (const float*)d_in[3];
    const float* bq    = (const float*)d_in[4];
    const float* Wk    = (const float*)d_in[5];
    const float* bk    = (const float*)d_in[6];
    const float* Wc    = (const float*)d_in[7];
    const float* bc    = (const float*)d_in[8];
    float* out = (float*)d_out;

    _Float16* Qbig = (_Float16*)d_ws;                       // 8 MB
    _Float16* Kbig = Qbig + (size_t)4 * 1024 * 1024;        // 8 MB

    dim3 g1(4 * 1024, 2);
    proj_tanh_kernel<<<g1, 256, 0, stream>>>(query, keyin, Wq, bq, Wk, bk, Wc,
                                             Qbig, Kbig);

    dim3 g2(64, 4);
    // 160 KB dynamic LDS: 8 waves x 2 x 10240 B B-tiles (comb tile aliased)
    attn_softmax_kernel<<<g2, 256, 163840, stream>>>(Qbig, Kbig, mask, bc, out);
}
